// VectorQuantizerEMA_38878043963541
// MI455X (gfx1250) — matrix operand order repaired and checked by reading, not compiled, not run on hardware
//
#include <hip/hip_runtime.h>
#include <hip/hip_bf16.h>
#include <math.h>

typedef float v8f   __attribute__((ext_vector_type(8)));
typedef __bf16 v16bf __attribute__((ext_vector_type(16)));
typedef unsigned int u32x4 __attribute__((ext_vector_type(4)));
typedef int i32x4 __attribute__((ext_vector_type(4)));
typedef int i32x8 __attribute__((ext_vector_type(8)));

// amdgpu-toolchain (clang-23 + therock headers) uses the 6-arg TDM builtin and
// is the only toolchain shipping this header; ROCm 7.2 (clang-22) is 5-arg.
#if __has_include(<hip/amd_detail/amd_gfx1250_TDM.h>)
#define TDM_SIX_ARG 1
#endif

constexpr int D = 256;
constexpr int K = 8192;
constexpr int N = 16384;
constexpr int CD = 512;             // split contraction length (xh|xl, eh|el)
constexpr float DECAYC = 0.99f;
constexpr float EPSV   = 1e-5f;
constexpr float COMMIT = 0.25f;

// ---- workspace layout ----
// float offsets:
constexpr size_t IDX_OFF = 0;                  // int[N] argmin indices
constexpr size_t CS_OFF  = 16384;              // float[K] cluster_size
constexpr size_t SC_OFF  = CS_OFF + K;         // [0]=loss accum, [1]=n_sum
constexpr size_t ESQ_OFF = SC_OFF + 2;         // float[K] ||e_k||^2
constexpr size_t DW_OFF  = 32800;              // float[D*K] dw scatter (phase 2)
// byte offsets (phase-1 bf16 panels OVERLAY the dw region; dw is zeroed only
// after the argmin kernel has consumed the panels):
constexpr size_t XHL_BYTE  = DW_OFF * 4;                          // bf16[N*512]
constexpr size_t EBFT_BYTE = XHL_BYTE + (size_t)N * CD * 2;       // bf16[K*512]

// ---- output layout (float offsets, concatenated tuple) ----
constexpr size_t Q_OFF    = 0;                 // quantized_st [N*D]
constexpr size_t LOSS_OFF = (size_t)N * D;     // scalar
constexpr size_t PERP_OFF = LOSS_OFF + 1;      // scalar
constexpr size_t ENC_OFF  = PERP_OFF + 1;      // indices as float [N]
constexpr size_t EMB_OFF  = ENC_OFF + N;       // new_embeddings [D*K]

// ------------------------------------------------------------------
// TDM 2-D tile load (global row-major -> LDS) with per-row DWORD padding,
// which realizes the bank-conflict-free LDS row strides in the DMA itself.
//   dsz_code: 0=1B 1=2B 2=4B 3=8B elements
//   pad_int_code: 0=2,...,7=256 dwords between pads; pad_amt_code: 0=1..127=128
// Issued by one wave; tracked on TENSORcnt.
// ------------------------------------------------------------------
__device__ __forceinline__ void tdm_load_2d_to_lds(
    unsigned lds_byte_off, const void* gptr, unsigned dsz_code,
    unsigned tile_d0, unsigned tile_d1,
    unsigned tensor_d0, unsigned tensor_d1,
    unsigned long long stride0_elems,
    unsigned pad_int_code, unsigned pad_amt_code, bool pad_en) {
  unsigned long long ga = (unsigned long long)(uintptr_t)gptr;
  u32x4 g0;
  g0.x = 1u;                                     // count=1, user mode, no gather
  g0.y = lds_byte_off;                           // lds_addr
  g0.z = (unsigned)ga;                           // global_addr[31:0]
  g0.w = (unsigned)((ga >> 32) & 0x01FFFFFFull)  // global_addr[56:32]
         | 0x80000000u;                          // type=2 ("image")
  i32x8 g1;
  unsigned w0 = (dsz_code << 16);
  if (pad_en)
    w0 |= (1u << 20) | (pad_int_code << 22) | (pad_amt_code << 25);
  g1[0] = (int)w0;                               // wg_mask=0 (not in cluster)
  g1[1] = (int)((tensor_d0 & 0xFFFFu) << 16);
  g1[2] = (int)((tensor_d0 >> 16) | ((tensor_d1 & 0xFFFFu) << 16));
  g1[3] = (int)((tensor_d1 >> 16) | (tile_d0 << 16));
  g1[4] = (int)(tile_d1 & 0xFFFFu);              // tile_dim2=0 -> 2D tile
  g1[5] = (int)(unsigned)(stride0_elems & 0xFFFFFFFFull);
  g1[6] = (int)(unsigned)((stride0_elems >> 32) & 0xFFFFull);
  g1[7] = 0;
  i32x4 z4 = {0, 0, 0, 0};
#ifdef TDM_SIX_ARG
  i32x8 z8 = {0, 0, 0, 0, 0, 0, 0, 0};
  __builtin_amdgcn_tensor_load_to_lds(g0, g1, z4, z4, z8, 0);
#else
  __builtin_amdgcn_tensor_load_to_lds(g0, g1, z4, z4, 0);
#endif
}

// ------------------------------------------------------------------
// Split x into bf16 hi/lo panels: xhl[n][0..255]=hi, [256..511]=lo.
__global__ void __launch_bounds__(256)
prep_x_kernel(const float* __restrict__ x, unsigned short* __restrict__ xhl_u) {
  __bf16* xhl = (__bf16*)xhl_u;
  const int n = blockIdx.x, t = threadIdx.x;
  const float v = x[(size_t)n * D + t];
  const __bf16 hi = (__bf16)v;
  const __bf16 lo = (__bf16)(v - (float)hi);
  xhl[(size_t)n * CD + t]      = hi;
  xhl[(size_t)n * CD + D + t]  = lo;
}

// Split embeddings, TRANSPOSED to k-major (ebfT[k][0..255]=eh, [256..511]=el)
// so the WMMA B-panel is a plain row-major TDM tile and B-fragments are
// contiguous 16B LDS loads. Fuses the ||e_k||^2 reduction.
__global__ void __launch_bounds__(256)
prep_e_kernel(const float* __restrict__ emb, unsigned short* __restrict__ ebfT_u,
              float* __restrict__ ws) {
  __shared__ float red[256];
  __bf16* ebfT = (__bf16*)ebfT_u;
  const int k = blockIdx.x, t = threadIdx.x;   // t = d
  const float v = emb[(size_t)t * K + k];
  const __bf16 hi = (__bf16)v;
  const __bf16 lo = (__bf16)(v - (float)hi);
  ebfT[(size_t)k * CD + t]     = hi;
  ebfT[(size_t)k * CD + D + t] = lo;
  red[t] = v * v;
  __syncthreads();
  for (int s = 128; s > 0; s >>= 1) {
    if (t < s) red[t] += red[t + s];
    __syncthreads();
  }
  if (t == 0) ws[ESQ_OFF + k] = red[0];
}

// ------------------------------------------------------------------
// bf16x3 WMMA distance/argmax kernel.
// dot(x,e) ~= xh.eh + xh.el + xl.eh  (error ~2^-16, argmin-safe), realized as
// 24 k-steps of v_wmma_f32_16x16x32_bf16 per 16x16 tile (vs 64 fp32 steps).
// Block = 256 threads (8 waves), 128 rows per block.
// A panel 128x512 bf16 and double-buffered B panels 64x512 bf16 are DMA'd by
// the TDM with 4-dword row pads (row stride 1040 B, 16B-aligned), overlapping
// TDM load(chunk kc+1) with compute(chunk kc); wave 0 drains TENSORcnt.
// ------------------------------------------------------------------
constexpr int ROWS = 128;                 // A rows per block
constexpr int RSTB = 1040;                // LDS row stride bytes (256+4 dwords)
constexpr int NK   = 64;                  // K-chunk width (columns)
constexpr size_t A_BYTES = (size_t)ROWS * RSTB;  // 133120
constexpr size_t B_BYTES = (size_t)NK * RSTB;    //  66560

union Frag {
  u32x4 q[2];
  v16bf v;
};

__global__ void __launch_bounds__(256)
argmin_wmma_kernel(const unsigned short* __restrict__ xhl_u,
                   const unsigned short* __restrict__ ebfT_u,
                   const float* __restrict__ ws_ro, int* __restrict__ idx_out,
                   float* __restrict__ enc_out) {
  extern __shared__ float smem[];
  char* lds = (char*)smem;
  const float* esq_g = ws_ro + ESQ_OFF;

  const int t    = threadIdx.x;
  const int base = blockIdx.x * ROWS;
  const int wave = t >> 5;
  const int lane = t & 31;
  const int m    = lane & 15;   // row within 16-tile (A) == col within tile (B/C)
  const int h    = lane >> 4;   // half-wave selector in bf16 fragment layouts

  // Prologue: DMA the A slab and the first B panel; drain and publish.
  if (wave == 0) {
    tdm_load_2d_to_lds(0u, xhl_u + (size_t)base * CD, /*dsz=2B*/ 1,
                       /*tile*/ CD, ROWS, /*tensor*/ CD, N,
                       /*stride0*/ CD, /*pad*/ 7, 3, true);
    tdm_load_2d_to_lds((unsigned)A_BYTES, ebfT_u, 1,
                       CD, NK, CD, K, CD, 7, 3, true);
    __builtin_amdgcn_s_wait_tensorcnt(0);
  }
  __syncthreads();

  const char* Arow = lds + (size_t)(wave * 16 + m) * RSTB;

  float bestv[8];
  int   besti[8];
#pragma unroll
  for (int j = 0; j < 8; ++j) { bestv[j] = -3.4e38f; besti[j] = 0; }

  constexpr int NCH = K / NK;
  for (int kc = 0; kc < NCH; ++kc) {
    const int k0 = kc * NK;
    const char* Bbase = lds + A_BYTES + (size_t)(kc & 1) * B_BYTES;

    // Prefetch next panel via TDM; overlaps with the WMMA loop below.
    if (wave == 0 && (kc + 1) < NCH) {
      tdm_load_2d_to_lds((unsigned)(A_BYTES + (size_t)((kc + 1) & 1) * B_BYTES),
                         ebfT_u + (size_t)(k0 + NK) * CD, 1,
                         CD, NK, CD, K, CD, 7, 3, true);
    }

    const char* Brow[4];
#pragma unroll
    for (int kt = 0; kt < 4; ++kt) Brow[kt] = Bbase + (size_t)(kt * 16 + m) * RSTB;

    float he[4];
#pragma unroll
    for (int kt = 0; kt < 4; ++kt) he[kt] = 0.5f * esq_g[k0 + kt * 16 + m];

    v8f acc[4] = {};
    // 8 windows x {xh.eh + xh.el + xl.eh}; B-hi fragment shared by two passes.
#pragma unroll
    for (int j = 0; j < 8; ++j) {
      const int aw = j * 64 + 16 * h;     // A window byte offset
      Frag Ah, Al;
      Ah.q[0] = *(const u32x4*)(Arow + aw);
      Ah.q[1] = *(const u32x4*)(Arow + aw + 32);
      Al.q[0] = *(const u32x4*)(Arow + 512 + aw);
      Al.q[1] = *(const u32x4*)(Arow + 512 + aw + 32);
      const int bw = j * 64 + 16 * h;     // B window byte offset
#pragma unroll
      for (int kt = 0; kt < 4; ++kt) {
        Frag Bh, Bl;
        Bh.q[0] = *(const u32x4*)(Brow[kt] + bw);
        Bh.q[1] = *(const u32x4*)(Brow[kt] + bw + 32);
        Bl.q[0] = *(const u32x4*)(Brow[kt] + 512 + bw);
        Bl.q[1] = *(const u32x4*)(Brow[kt] + 512 + bw + 32);
        acc[kt] = __builtin_amdgcn_wmma_f32_16x16x32_bf16(
            false, Ah.v, false, Bh.v, (short)0, acc[kt], false, false);
        acc[kt] = __builtin_amdgcn_wmma_f32_16x16x32_bf16(
            false, Ah.v, false, Bl.v, (short)0, acc[kt], false, false);
        acc[kt] = __builtin_amdgcn_wmma_f32_16x16x32_bf16(
            false, Al.v, false, Bh.v, (short)0, acc[kt], false, false);
      }
    }

    // Fold chunk into the running argmax.
#pragma unroll
    for (int kt = 0; kt < 4; ++kt) {
      const int col = k0 + kt * 16 + m;
#pragma unroll
      for (int j = 0; j < 8; ++j) {
        float s = acc[kt][j] - he[kt];
        if (s > bestv[j]) { bestv[j] = s; besti[j] = col; }
      }
    }

    __syncthreads();                      // everyone done reading this panel
    if (wave == 0) __builtin_amdgcn_s_wait_tensorcnt(0);  // next panel landed
    __syncthreads();                      // publish next panel to all waves
  }

  // Cross-lane reduction: row r of a wave tile lives in lanes 0-15 (slot r)
  // and lanes 16-31 (slot r-8). Reuse the B region as scratch.
  float* Redv = (float*)(lds + A_BYTES);          // 8*32*8 floats
  int*   Redi = (int*)(Redv + 8 * 32 * 8);        // 8*32*8 ints
#pragma unroll
  for (int j = 0; j < 8; ++j) {
    Redv[(wave * 32 + lane) * 8 + j] = bestv[j];
    Redi[(wave * 32 + lane) * 8 + j] = besti[j];
  }
  __syncthreads();
  if (lane < 16) {
    const int r    = lane;
    const int lo   = (r < 8) ? 0 : 16;
    const int slot = (r < 8) ? r : (r - 8);
    float bv = -3.4e38f;
    int   bi = 0;
    for (int L = 0; L < 16; ++L) {
      const int e = (wave * 32 + lo + L) * 8 + slot;
      float v = Redv[e];
      if (v > bv) { bv = v; bi = Redi[e]; }
    }
    const int row = base + wave * 16 + r;
    idx_out[row] = bi;
    enc_out[row] = (float)bi;
  }
}

// ------------------------------------------------------------------
// Zero cluster_size + scalars + dw (runs AFTER argmin: dw overlays panels).
__global__ void zero_ws_kernel(float* __restrict__ ws) {
  size_t i = (size_t)blockIdx.x * blockDim.x + threadIdx.x;
  constexpr size_t NZ1 = (size_t)K + 2;   // CS + scalars (contiguous)
  if (i < NZ1) {
    ws[CS_OFF + i] = 0.0f;
  } else {
    size_t j = i - NZ1;
    if (j < (size_t)D * K) ws[DW_OFF + j] = 0.0f;
  }
}

// ------------------------------------------------------------------
// Gather codewords, commitment-loss partials, histogram, dw scatter.
__global__ void __launch_bounds__(256)
gather_stats_kernel(const float* __restrict__ x, const float* __restrict__ emb,
                    const int* __restrict__ idx, float* __restrict__ qout,
                    float* __restrict__ ws) {
  __shared__ float red[256];
  const int n  = blockIdx.x;
  const int t  = threadIdx.x;
  const int id = idx[n];
  const float q  = emb[(size_t)t * K + id];
  const float xv = x[(size_t)n * D + t];
  qout[(size_t)n * D + t] = q;     // straight-through output == quantized
  const float dd = q - xv;
  red[t] = dd * dd;
  atomicAdd(&ws[DW_OFF + (size_t)t * K + id], xv);   // dw[d, idx] += x
  __syncthreads();
  for (int s = 128; s > 0; s >>= 1) {
    if (t < s) red[t] += red[t + s];
    __syncthreads();
  }
  if (t == 0) {
    atomicAdd(&ws[SC_OFF + 0], red[0]);
    atomicAdd(&ws[CS_OFF + id], 1.0f);
  }
}

// ------------------------------------------------------------------
__global__ void __launch_bounds__(256)
finalize_scalars_kernel(const float* __restrict__ ch, const int* __restrict__ counter,
                        float* __restrict__ ws, float* __restrict__ out) {
  __shared__ float r1[256], r2[256];
  const int t = threadIdx.x;
  const float bias = 1.0f - powf(DECAYC, (float)(counter[0] + 1));
  float nsum = 0.0f, plog = 0.0f;
  for (int k = t; k < K; k += 256) {
    const float cs = ws[CS_OFF + k];
    const float hc = ch[k] * DECAYC + cs * (1.0f - DECAYC);
    nsum += hc / bias;
    const float p = cs / (float)N;
    plog += p * logf(p + 1e-10f);
  }
  r1[t] = nsum; r2[t] = plog;
  __syncthreads();
  for (int s = 128; s > 0; s >>= 1) {
    if (t < s) { r1[t] += r1[t + s]; r2[t] += r2[t + s]; }
    __syncthreads();
  }
  if (t == 0) {
    ws[SC_OFF + 1] = r1[0];
    out[LOSS_OFF] = COMMIT * ws[SC_OFF + 0] / (float)((size_t)N * D);
    out[PERP_OFF] = expf(-r2[0]);
  }
}

// ------------------------------------------------------------------
__global__ void __launch_bounds__(256)
new_embeddings_kernel(const float* __restrict__ ch, const float* __restrict__ dwh,
                      const int* __restrict__ counter, const float* __restrict__ ws,
                      float* __restrict__ out) {
  const size_t e = (size_t)blockIdx.x * 256 + threadIdx.x;
  const int k = (int)(e & (size_t)(K - 1));
  const float bias  = 1.0f - powf(DECAYC, (float)(counter[0] + 1));
  const float nsum  = ws[SC_OFF + 1];
  const float cs    = ws[CS_OFF + k];
  const float avg_c = (ch[k] * DECAYC + cs * (1.0f - DECAYC)) / bias;
  const float norm_c = (avg_c + EPSV) / (nsum + (float)K * EPSV) * nsum;
  const float avg_d = (dwh[e] * DECAYC + ws[DW_OFF + e] * (1.0f - DECAYC)) / bias;
  out[EMB_OFF + e] = avg_d / norm_c;
}

// ------------------------------------------------------------------
extern "C" void kernel_launch(void* const* d_in, const int* in_sizes, int n_in,
                              void* d_out, int out_size, void* d_ws, size_t ws_size,
                              hipStream_t stream) {
  const float* x       = (const float*)d_in[0];
  const float* emb     = (const float*)d_in[1];
  const float* ch      = (const float*)d_in[2];
  const float* dwh     = (const float*)d_in[3];
  const int*   counter = (const int*)d_in[4];
  float* out = (float*)d_out;
  float* ws  = (float*)d_ws;
  int* idx_ws = (int*)d_ws;  // IDX_OFF == 0
  unsigned short* xhl  = (unsigned short*)((char*)d_ws + XHL_BYTE);
  unsigned short* ebfT = (unsigned short*)((char*)d_ws + EBFT_BYTE);

  // Phase 1: split-precision panels + esq, then WMMA argmin.
  prep_x_kernel<<<N, 256, 0, stream>>>(x, xhl);
  prep_e_kernel<<<K, 256, 0, stream>>>(emb, ebfT, ws);
  {
    size_t shmem = A_BYTES + 2 * B_BYTES;   // 266240 B of the 320KB WGP LDS
    argmin_wmma_kernel<<<N / ROWS, 256, shmem, stream>>>(xhl, ebfT, ws, idx_ws,
                                                         out + ENC_OFF);
  }
  // Phase 2: panels dead; zero the overlaying dw buffer + stats, then EMA.
  {
    size_t nz = (size_t)K + 2 + (size_t)D * K;
    int blocks = (int)((nz + 255) / 256);
    zero_ws_kernel<<<blocks, 256, 0, stream>>>(ws);
  }
  gather_stats_kernel<<<N, 256, 0, stream>>>(x, emb, idx_ws, out + Q_OFF, ws);
  finalize_scalars_kernel<<<1, 256, 0, stream>>>(ch, counter, ws, out);
  new_embeddings_kernel<<<(int)(((size_t)D * K) / 256), 256, 0, stream>>>(
      ch, dwh, counter, ws, out);
}
